// PFASModule_28741921144997
// MI455X (gfx1250) — matrix operand order repaired and checked by reading, not compiled, not run on hardware
//
#include <hip/hip_runtime.h>
#include <hip/hip_bf16.h>
#include <math.h>

typedef __attribute__((ext_vector_type(16))) _Float16 v16h;
typedef __attribute__((ext_vector_type(8)))  float    v8f;

#define GK 32          // neighbors kept (reference K)
#define GBLK 256       // threads per geometry block (= query points per block)

// ---------------------------------------------------------------------------
// Kernel 1: per-batch KNN + covariance eigenvalues -> linearity, density.
// Whole batch point cloud (72KB) lives in LDS (CDNA5: 320KB/WGP).
// Per-thread top-32 candidate list in LDS, slot-major for conflict-free rescan.
// ---------------------------------------------------------------------------
__global__ __launch_bounds__(GBLK)
void geom_kernel(const float* __restrict__ coord,
                 float* __restrict__ lin_out, float* __restrict__ den_out,
                 int P, int blocksPerBatch) {
    extern __shared__ float smem[];
    float* c    = smem;                       // [P*3] batch coordinates
    float* d2h  = c + (size_t)P * 3;          // [GK][GBLK] candidate d2, slot-major
    int*   idxh = (int*)(d2h + GK * GBLK);    // [GK][GBLK] candidate local index

    const int tid    = threadIdx.x;
    const int b      = blockIdx.x / blocksPerBatch;
    const int chunk  = blockIdx.x % blocksPerBatch;
    const int qLocal = chunk * GBLK + tid;    // local point index within batch

    // Stage the batch's coordinates into LDS (coalesced), init candidate slots.
    const float* cb = coord + (size_t)b * P * 3;
    for (int i = tid; i < P * 3; i += GBLK) {
        __builtin_prefetch(cb + i + 4 * GBLK, 0, 1);   // global_prefetch_b8
        c[i] = cb[i];
    }
    #pragma unroll
    for (int s = 0; s < GK; ++s) {
        d2h[s * GBLK + tid]  = 3.0e38f;
        idxh[s * GBLK + tid] = 0;
    }
    __syncthreads();

    const float qx = c[qLocal * 3 + 0];
    const float qy = c[qLocal * 3 + 1];
    const float qz = c[qLocal * 3 + 2];

    float curMax = 3.0e38f;   // current worst (largest) kept distance
    int   curSlot = 0;

    for (int j = 0; j < P; ++j) {
        // all lanes read the same candidate -> LDS broadcast, no conflicts
        const float dx = c[j * 3 + 0] - qx;
        const float dy = c[j * 3 + 1] - qy;
        const float dz = c[j * 3 + 2] - qz;
        const float d2 = dx * dx + dy * dy + dz * dz;
        if (j != qLocal && d2 < curMax) {
            d2h[curSlot * GBLK + tid]  = d2;
            idxh[curSlot * GBLK + tid] = j;
            // rescan for new max (rare: ~170x per thread over 6144 candidates)
            float m = -1.0f; int ms = 0;
            #pragma unroll
            for (int s = 0; s < GK; ++s) {
                const float v = d2h[s * GBLK + tid];
                if (v > m) { m = v; ms = s; }
            }
            curMax = m; curSlot = ms;
        }
    }

    // density = 1 / (mean neighbor distance + 1e-6)
    float dsum = 0.0f;
    #pragma unroll
    for (int s = 0; s < GK; ++s)
        dsum += sqrtf(fmaxf(d2h[s * GBLK + tid], 1e-12f));
    const float density = 1.0f / (dsum * (1.0f / GK) + 1e-6f);

    // covariance of neighbor coordinates (moment form)
    float sx = 0, sy = 0, sz = 0;
    float sxx = 0, sxy = 0, sxz = 0, syy = 0, syz = 0, szz = 0;
    #pragma unroll
    for (int s = 0; s < GK; ++s) {
        const int j = idxh[s * GBLK + tid];
        const float x = c[j * 3 + 0], y = c[j * 3 + 1], z = c[j * 3 + 2];
        sx += x; sy += y; sz += z;
        sxx += x * x; sxy += x * y; sxz += x * z;
        syy += y * y; syz += y * z; szz += z * z;
    }
    const float invK = 1.0f / GK;
    const float mx = sx * invK, my = sy * invK, mz = sz * invK;
    const float invKm1 = 1.0f / (GK - 1);
    const float a00 = (sxx - GK * mx * mx) * invKm1;
    const float a01 = (sxy - GK * mx * my) * invKm1;
    const float a02 = (sxz - GK * mx * mz) * invKm1;
    const float a11 = (syy - GK * my * my) * invKm1;
    const float a12 = (syz - GK * my * mz) * invKm1;
    const float a22 = (szz - GK * mz * mz) * invKm1;

    // closed-form symmetric 3x3 eigenvalues (descending e0 >= e1 >= e2)
    const float p1 = a01 * a01 + a02 * a02 + a12 * a12;
    const float q  = (a00 + a11 + a22) * (1.0f / 3.0f);
    const float t0 = a00 - q, t1 = a11 - q, t2 = a22 - q;
    const float p2 = t0 * t0 + t1 * t1 + t2 * t2 + 2.0f * p1;
    const float p  = sqrtf(p2 * (1.0f / 6.0f));
    float e0, e1, e2;
    if (p < 1e-20f) {
        e0 = e1 = e2 = q;
    } else {
        const float ip = 1.0f / p;
        const float b00 = t0 * ip, b11 = t1 * ip, b22 = t2 * ip;
        const float b01 = a01 * ip, b02 = a02 * ip, b12 = a12 * ip;
        float r = 0.5f * (b00 * (b11 * b22 - b12 * b12)
                        - b01 * (b01 * b22 - b12 * b02)
                        + b02 * (b01 * b12 - b11 * b02));
        r = fminf(1.0f, fmaxf(-1.0f, r));
        const float phi = acosf(r) * (1.0f / 3.0f);
        e0 = q + 2.0f * p * cosf(phi);
        e2 = q + 2.0f * p * cosf(phi + 2.0943951023931953f);
        e1 = 3.0f * q - e0 - e2;
    }
    const float esum = e0 + e1 + e2;
    const float linearity = (e0 - e1 - e2) / esum;

    const int gq = b * P + qLocal;
    lin_out[gq] = linearity;
    den_out[gq] = density;
}

// ---------------------------------------------------------------------------
// Kernel 2: fused MLP via v_wmma_f32_16x16x32_f16 + BN + ReLU + tiny GEMM +
// softmax + blend with linearity/density.
// Block = 256 threads = 8 waves; each wave computes a 16x64 tile of h.
// ---------------------------------------------------------------------------
__global__ __launch_bounds__(256)
void mlp_kernel(const float* __restrict__ feat, const float* __restrict__ W1,
                const float* __restrict__ b1, const float* __restrict__ gamma,
                const float* __restrict__ beta, const float* __restrict__ bn_mean,
                const float* __restrict__ bn_var, const float* __restrict__ W2,
                const float* __restrict__ b2,
                const float* __restrict__ lin, const float* __restrict__ den,
                float* __restrict__ out, int N) {
    __shared__ alignas(32) _Float16 w1t[64 * 64];   // W1 transposed: [n][k], f16
    __shared__ float scale_s[64], bias_s[64];       // folded BN affine
    __shared__ float w2s[64 * 3];
    __shared__ float b2s[3];
    __shared__ float hbuf[8][16][64];               // per-wave h staging

    const int tid = threadIdx.x;

    // Stage W1 (transposed, f16) and the folded BN parameters.
    for (int i = tid; i < 64 * 64; i += 256) {
        const int k = i >> 6, n = i & 63;           // coalesced read of W1[k][n]
        w1t[n * 64 + k] = (_Float16)W1[i];
    }
    if (tid < 64) {
        const float s = gamma[tid] * rsqrtf(bn_var[tid] + 1e-5f);
        scale_s[tid] = s;
        bias_s[tid]  = s * (b1[tid] - bn_mean[tid]) + beta[tid];
    }
    if (tid < 192) w2s[tid] = W2[tid];
    if (tid < 3)   b2s[tid] = b2[tid];
    __syncthreads();

    const int wave = tid >> 5;
    const int lane = tid & 31;
    const int n    = lane & 15;      // N-column / A-row within tile
    const int hi   = lane >> 4;      // half-wave selector for K sub-ranges
    const int rowBase = (blockIdx.x * 8 + wave) * 16;
    const float* fr = feat + (size_t)(rowBase + n) * 64;

    v8f acc[4] = {v8f{}, v8f{}, v8f{}, v8f{}};

    #pragma unroll
    for (int s = 0; s < 2; ++s) {                    // K = 64 -> 2 steps of 32
        const int kbase = s * 32;
        // A fragment: lane holds M = lane%16;
        // element e -> K = kbase + hi*16 + e (matches B)
        v16h a;
        {
            const int k1 = kbase + hi * 16;
            const float4 f0 = *(const float4*)(fr + k1);
            const float4 f1 = *(const float4*)(fr + k1 + 4);
            const int k2 = kbase + hi * 16 + 8;
            const float4 f2 = *(const float4*)(fr + k2);
            const float4 f3 = *(const float4*)(fr + k2 + 4);
            a[0] = (_Float16)f0.x; a[1] = (_Float16)f0.y;
            a[2] = (_Float16)f0.z; a[3] = (_Float16)f0.w;
            a[4] = (_Float16)f1.x; a[5] = (_Float16)f1.y;
            a[6] = (_Float16)f1.z; a[7] = (_Float16)f1.w;
            a[8]  = (_Float16)f2.x; a[9]  = (_Float16)f2.y;
            a[10] = (_Float16)f2.z; a[11] = (_Float16)f2.w;
            a[12] = (_Float16)f3.x; a[13] = (_Float16)f3.y;
            a[14] = (_Float16)f3.z; a[15] = (_Float16)f3.w;
        }
        #pragma unroll
        for (int t = 0; t < 4; ++t) {
            // B fragment: lane holds N = lane%16; element e -> K = kbase + hi*16 + e.
            // Contiguous in w1t -> one 32B LDS read.
            const int col = t * 16 + n;
            const v16h bfrag = *(const v16h*)&w1t[col * 64 + kbase + hi * 16];
            acc[t] = __builtin_amdgcn_wmma_f32_16x16x32_f16(
                false, a, false, bfrag, (short)0, acc[t], false, false);
        }
    }

    // Epilogue: folded BN + ReLU, stage h tile to LDS.
    // C/D layout: VGPR r holds row (r + 8*hi), column n.
    #pragma unroll
    for (int t = 0; t < 4; ++t) {
        const int col = t * 16 + n;
        const float sc = scale_s[col], bi = bias_s[col];
        #pragma unroll
        for (int r = 0; r < 8; ++r) {
            const int m = r + hi * 8;
            hbuf[wave][m][col] = fmaxf(0.0f, sc * acc[t][r] + bi);
        }
    }
    __syncthreads();

    // Lanes 0..15: one row each — 64x3 GEMM, softmax, blend, store.
    if (lane < 16) {
        const int row = rowBase + lane;
        if (row < N) {
            float l0 = b2s[0], l1 = b2s[1], l2 = b2s[2];
            #pragma unroll 8
            for (int ci = 0; ci < 64; ++ci) {
                const float hv = hbuf[wave][lane][ci];
                l0 += hv * w2s[ci * 3 + 0];
                l1 += hv * w2s[ci * 3 + 1];
                l2 += hv * w2s[ci * 3 + 2];
            }
            const float mxl = fmaxf(l0, fmaxf(l1, l2));
            const float x0 = __expf(l0 - mxl);
            const float x1 = __expf(l1 - mxl);
            const float x2 = __expf(l2 - mxl);
            const float inv = 1.0f / (x0 + x1 + x2);
            const float p0 = x0 * inv, p1 = x1 * inv, p2 = x2 * inv;

            const float li = lin[row], de = den[row];
            const float tower = (de * 2.0f + p0) * (1.0f / 3.0f);
            const float bg    = (fmaxf(1.0f - li, 1.0f - de) + p1) * (1.0f / 3.0f);
            const float line  = (li * 2.0f + p2) * (1.0f / 3.0f);
            // grids: tower [.1,.1,.1], bg [.5,.5,.5], line [.2,.2,.25]
            const float gxy = tower * 0.1f + bg * 0.5f + line * 0.2f  + 1e-6f;
            const float gz  = tower * 0.1f + bg * 0.5f + line * 0.25f + 1e-6f;
            out[row * 3 + 0] = gxy;
            out[row * 3 + 1] = gxy;
            out[row * 3 + 2] = gz;
        }
    }
}

// ---------------------------------------------------------------------------
extern "C" void kernel_launch(void* const* d_in, const int* in_sizes, int n_in,
                              void* d_out, int out_size, void* d_ws, size_t ws_size,
                              hipStream_t stream) {
    const float* feat    = (const float*)d_in[0];
    const float* coord   = (const float*)d_in[1];
    // d_in[2] = batch ids (layout is fixed: B consecutive groups of P) — unused
    const float* W1      = (const float*)d_in[3];
    const float* b1      = (const float*)d_in[4];
    const float* gamma   = (const float*)d_in[5];
    const float* beta    = (const float*)d_in[6];
    const float* bn_mean = (const float*)d_in[7];
    const float* bn_var  = (const float*)d_in[8];
    const float* W2      = (const float*)d_in[9];
    const float* b2      = (const float*)d_in[10];
    float* out = (float*)d_out;

    const int N = in_sizes[0] / 64;   // C = 64
    const int B = 4;
    const int P = N / B;

    float* lin = (float*)d_ws;        // [N]
    float* den = lin + N;             // [N]

    const int bpb = P / GBLK;         // geometry blocks per batch
    const size_t shmem = (size_t)P * 3 * sizeof(float)          // coords
                       + (size_t)GK * GBLK * sizeof(float)      // d2 slots
                       + (size_t)GK * GBLK * sizeof(int);       // idx slots
    geom_kernel<<<dim3(B * bpb), dim3(GBLK), shmem, stream>>>(coord, lin, den, P, bpb);

    mlp_kernel<<<dim3(N / 128), dim3(256), 0, stream>>>(
        feat, W1, b1, gamma, beta, bn_mean, bn_var, W2, b2, lin, den, out, N);
}
